// ActionConditionedFactorizedTransition_77567109366249
// MI455X (gfx1250) — compile-verified, matrix operand order checked by reading, not run on hardware
//
#include <hip/hip_runtime.h>
#include <stdint.h>

// Problem constants (match reference)
#define EMB 128
#define HID 128
#define BATCH 16
#define KSP 128
#define TILE 128          // s-tile width
#define LH 136            // LDS stride in halves (even, 16B-compatible padding)
#define SST 132           // padded stride for wave-partial array
#define NEG_INF (-3.402823466e38f)
#define INV_SQRT_H 0.08838834764831845f

typedef __attribute__((ext_vector_type(16))) _Float16 v16h;
typedef __attribute__((ext_vector_type(8)))  float    v8f;

// ---------------------------------------------------------------------------
// Assemble a 16x16x32 WMMA A/B fragment for one 32-wide K-chunk from an LDS
// row (row-major [row][e], stride LH halves). ISA layout (CDNA5 ISA):
//   lanes 0-15 : elements K = hc*32 + {0..7, 16..23}
//   lanes 16-31: elements K = hc*32 + {8..15, 24..31}
// ---------------------------------------------------------------------------
__device__ __forceinline__ v16h load_frag(const _Float16* row, int hc, int lane) {
    const uint32_t* p = (const uint32_t*)row;
    int pb = hc * 16 + ((lane & 16) ? 4 : 0);
    union { v16h v; uint32_t u[8]; } f;
#pragma unroll
    for (int i = 0; i < 4; ++i) {
        f.u[i]     = p[pb + i];
        f.u[4 + i] = p[pb + 8 + i];
    }
    return f.v;
}

// ---------------------------------------------------------------------------
// One wave computes a 16x128 strip: 8 accumulator tiles over K=128.
// hc outer / nt inner: A-fragment loaded once per K-chunk, 8 independent
// WMMAs per chunk (hides WMMA->WMMA hazard latency).
// ---------------------------------------------------------------------------
__device__ __forceinline__ void mma_row_strip(const _Float16* aRow,
                                              const _Float16* bBase,
                                              int lane, v8f acc[8]) {
    const _Float16* brow = bBase + (size_t)(lane & 15) * LH;
#pragma unroll
    for (int hc = 0; hc < 4; ++hc) {
        v16h a = load_frag(aRow, hc, lane);
#pragma unroll
        for (int nt = 0; nt < 8; ++nt) {
            v16h bf = load_frag(brow + (size_t)nt * 16 * LH, hc, lane);
            acc[nt] = __builtin_amdgcn_wmma_f32_16x16x32_f16(
                false, a, false, bf, (short)0, acc[nt], false, false);
        }
    }
}

__device__ __forceinline__ uint32_t pack2h(float a, float b) {
    union { _Float16 h[2]; uint32_t u; } t;
    t.h[0] = (_Float16)a; t.h[1] = (_Float16)b;
    return t.u;
}

// Stage a [rows<=128][128] f32 matrix into LDS f16 [128][LH]; rows >= validRows -> 0.
__device__ __forceinline__ void stage_f32(const float* __restrict__ src,
                                          _Float16* dst, int validRows, int tid) {
    for (int i = tid; i < TILE * 32; i += 256) {
        int row = i >> 5, g = i & 31;                 // group of 4 floats
        float4 f = make_float4(0.f, 0.f, 0.f, 0.f);
        if (row < validRows) f = ((const float4*)src)[(size_t)row * 32 + g];
        uint2 u; u.x = pack2h(f.x, f.y); u.y = pack2h(f.z, f.w);
        *(uint2*)(dst + (size_t)row * LH + 4 * g) = u;
    }
}

// ===========================================================================
// Kernel 1: keys[s,h] = emb[s,:] @ Wk[h,:] + bk[h], stored f16.
// ===========================================================================
__global__ __launch_bounds__(256)
void k_keys(const float* __restrict__ emb, const float* __restrict__ Wk,
            const float* __restrict__ bk, _Float16* __restrict__ keysF16, int S) {
    __shared__ __align__(16) _Float16 aL[TILE * LH];   // emb tile [s][e]
    __shared__ __align__(16) _Float16 wL[HID * LH];    // Wk       [h][e]
    const int tid = threadIdx.x, lane = tid & 31, wave = tid >> 5;
    const int s0 = blockIdx.x * TILE;
    int vrows = S - s0; if (vrows > TILE) vrows = TILE;

    stage_f32(Wk, wL, HID, tid);
    stage_f32(emb + (size_t)s0 * EMB, aL, vrows, tid);
    __syncthreads();

    const int m0 = wave * 16, cb = lane & 15, half8 = (lane & 16) ? 8 : 0;
    const _Float16* arow = aL + (size_t)(m0 + cb) * LH;
    v8f acc[8] = {};
    mma_row_strip(arow, wL, lane, acc);

    const bool full = (s0 + TILE <= S);
#pragma unroll
    for (int nt = 0; nt < 8; ++nt) {
        const int col  = nt * 16 + cb;
        const float bv = bk[col];
        const int rb   = m0 + half8;
        if (full) {
#pragma unroll
            for (int r = 0; r < 8; ++r)
                keysF16[(size_t)(s0 + rb + r) * HID + col] = (_Float16)(acc[nt][r] + bv);
        } else {
#pragma unroll
            for (int r = 0; r < 8; ++r) {
                int s = s0 + rb + r;
                if (s < S) keysF16[(size_t)s * HID + col] = (_Float16)(acc[nt][r] + bv);
            }
        }
    }
}

// ===========================================================================
// Kernel 2: queries[b*K+k,h] = (emb[idcs[b,k],:] @ Wq[h,:] + bq[h]) / sqrt(H)
// ===========================================================================
__global__ __launch_bounds__(256)
void k_queries(const float* __restrict__ emb, const float* __restrict__ Wq,
               const float* __restrict__ bq, const int* __restrict__ idcs,
               _Float16* __restrict__ qF16) {
    __shared__ __align__(16) _Float16 aL[TILE * LH];
    __shared__ __align__(16) _Float16 wL[HID * LH];
    const int tid = threadIdx.x, lane = tid & 31, wave = tid >> 5;
    const int r0 = blockIdx.x * TILE;

    stage_f32(Wq, wL, HID, tid);
    for (int i = tid; i < TILE * 32; i += 256) {       // gathered A tile
        int row = i >> 5, g = i & 31;
        int sidx = idcs[r0 + row];
        float4 f = ((const float4*)(emb + (size_t)sidx * EMB))[g];
        uint2 u; u.x = pack2h(f.x, f.y); u.y = pack2h(f.z, f.w);
        *(uint2*)(aL + (size_t)row * LH + 4 * g) = u;
    }
    __syncthreads();

    const int m0 = wave * 16, cb = lane & 15, half8 = (lane & 16) ? 8 : 0;
    const _Float16* arow = aL + (size_t)(m0 + cb) * LH;
    v8f acc[8] = {};
    mma_row_strip(arow, wL, lane, acc);

#pragma unroll
    for (int nt = 0; nt < 8; ++nt) {
        const int col  = nt * 16 + cb;
        const float bv = bq[col];
        const int rb   = m0 + half8;
#pragma unroll
        for (int r = 0; r < 8; ++r)
            qF16[(size_t)(r0 + rb + r) * HID + col] =
                (_Float16)((acc[nt][r] + bv) * INV_SQRT_H);
    }
}

// Stage q[b] (128x128 f16) and keys tile (tsv x 128 f16, zero-padded) into LDS.
__device__ __forceinline__ void stage_qk(const _Float16* __restrict__ qF16,
                                         const _Float16* __restrict__ keysF16,
                                         _Float16* qL, _Float16* kL,
                                         int b, int s0, int tsv, int tid) {
    const uint4* qg = (const uint4*)(qF16 + (size_t)b * KSP * HID);
    uint4* ql4 = (uint4*)qL;
    uint4* kl4 = (uint4*)kL;
    for (int i = tid; i < TILE * 16; i += 256) {       // 16 uint4 per row
        int row = i >> 4, g = i & 15;
        ql4[row * (LH / 8) + g] = qg[i];               // LH/8 = 17 uint4 stride
    }
    for (int i = tid; i < TILE * 16; i += 256) {
        int row = i >> 4, g = i & 15;
        uint4 v = make_uint4(0u, 0u, 0u, 0u);
        if (row < tsv)
            v = ((const uint4*)(keysF16 + (size_t)(s0 + row) * HID))[g];
        kl4[row * (LH / 8) + g] = v;
    }
}

// ===========================================================================
// Kernel 3 (pass A): per-(b,k) local max & sumexp per 128-wide s-tile.
// All reductions in registers + cross-lane shuffles (no score LDS round-trip).
// ===========================================================================
__global__ __launch_bounds__(256)
void k_stats(const _Float16* __restrict__ qF16, const _Float16* __restrict__ keysF16,
             float* __restrict__ pmax, float* __restrict__ psum,
             int S, int numTiles) {
    __shared__ __align__(16) _Float16 qL[KSP * LH];
    __shared__ __align__(16) _Float16 kL[TILE * LH];
    const int tid = threadIdx.x, lane = tid & 31, wave = tid >> 5;
    const int tile = blockIdx.x, b = blockIdx.y;
    const int s0 = tile * TILE;
    int tsv = S - s0; if (tsv > TILE) tsv = TILE;

    stage_qk(qF16, keysF16, qL, kL, b, s0, tsv, tid);
    __syncthreads();

    const int m0 = wave * 16, cb = lane & 15, half8 = (lane & 16) ? 8 : 0;
    const _Float16* arow = qL + (size_t)(m0 + cb) * LH;
    v8f acc[8] = {};
    mma_row_strip(arow, kL, lane, acc);

    // C layout: acc[nt][r] = score(row = m0 + half8 + r, col = nt*16 + cb).
    // Shuffle masks 1/2/4/8 never flip lane bit 4 -> reductions stay within
    // each 16-lane half (same row set).
#pragma unroll
    for (int r = 0; r < 8; ++r) {
        float m = NEG_INF;
#pragma unroll
        for (int nt = 0; nt < 8; ++nt) {
            int col = nt * 16 + cb;
            float v = (col < tsv) ? acc[nt][r] : NEG_INF;
            m = fmaxf(m, v);
        }
        m = fmaxf(m, __shfl_xor(m, 1));
        m = fmaxf(m, __shfl_xor(m, 2));
        m = fmaxf(m, __shfl_xor(m, 4));
        m = fmaxf(m, __shfl_xor(m, 8));

        float s = 0.0f;
#pragma unroll
        for (int nt = 0; nt < 8; ++nt) {
            int col = nt * 16 + cb;
            if (col < tsv) s += __expf(acc[nt][r] - m);
        }
        s += __shfl_xor(s, 1);
        s += __shfl_xor(s, 2);
        s += __shfl_xor(s, 4);
        s += __shfl_xor(s, 8);

        if (cb == 0) {
            int k = m0 + half8 + r;
            size_t o = (size_t)(b * KSP + k) * numTiles + tile;
            pmax[o] = m;
            psum[o] = s;
        }
    }
}

// ===========================================================================
// Kernel 4: merge partials -> rowmax, weight = belief / sumexp.
// ===========================================================================
__global__ __launch_bounds__(256)
void k_reduce(const float* __restrict__ pmax, const float* __restrict__ psum,
              const float* __restrict__ belief,
              float* __restrict__ rowmax, float* __restrict__ roww, int numTiles) {
    const int row = blockIdx.x * 256 + threadIdx.x;   // 0..2047
    const float* pm = pmax + (size_t)row * numTiles;
    const float* ps = psum + (size_t)row * numTiles;
    float m = NEG_INF;
    for (int t = 0; t < numTiles; ++t) m = fmaxf(m, pm[t]);
    float sum = 0.0f;
    for (int t = 0; t < numTiles; ++t) sum += ps[t] * __expf(pm[t] - m);
    rowmax[row] = m;
    roww[row]   = belief[row] / sum;
}

// ===========================================================================
// Kernel 5 (pass B): recompute scores, out[b,s] = sum_k w_k*exp(score - m_k).
// k-contraction done in registers per wave; cross-wave combine via small LDS.
// ===========================================================================
__global__ __launch_bounds__(256)
void k_out(const _Float16* __restrict__ qF16, const _Float16* __restrict__ keysF16,
           const float* __restrict__ rowmax, const float* __restrict__ roww,
           float* __restrict__ out, int S, int numTiles) {
    __shared__ __align__(16) _Float16 qL[KSP * LH];
    __shared__ __align__(16) _Float16 kL[TILE * LH];
    __shared__ float mS[KSP];
    __shared__ float wS[KSP];
    __shared__ float pW[8][SST];                      // per-wave column partials
    const int tid = threadIdx.x, lane = tid & 31, wave = tid >> 5;
    const int tile = blockIdx.x, b = blockIdx.y;
    const int s0 = tile * TILE;
    int tsv = S - s0; if (tsv > TILE) tsv = TILE;

    if (tid < KSP) {
        mS[tid] = rowmax[b * KSP + tid];
        wS[tid] = roww[b * KSP + tid];
    }
    stage_qk(qF16, keysF16, qL, kL, b, s0, tsv, tid);
    __syncthreads();

    const int m0 = wave * 16, cb = lane & 15, half8 = (lane & 16) ? 8 : 0;
    const _Float16* arow = qL + (size_t)(m0 + cb) * LH;
    v8f acc[8] = {};
    mma_row_strip(arow, kL, lane, acc);

    float mv[8], wv[8];
#pragma unroll
    for (int r = 0; r < 8; ++r) {
        int k = m0 + half8 + r;
        mv[r] = mS[k];
        wv[r] = wS[k];
    }
#pragma unroll
    for (int nt = 0; nt < 8; ++nt) {
        float p = 0.0f;
#pragma unroll
        for (int r = 0; r < 8; ++r)
            p += wv[r] * __expf(acc[nt][r] - mv[r]);
        p += __shfl_xor(p, 16);                       // combine row halves
        if (lane < 16) pW[wave][nt * 16 + lane] = p;  // 16-row partial per col
    }
    __syncthreads();

    if (tid < tsv) {
        float sum = 0.0f;
#pragma unroll
        for (int w = 0; w < 8; ++w) sum += pW[w][tid];
        out[(size_t)b * S + s0 + tid] = sum;
    }
}

// ===========================================================================
extern "C" void kernel_launch(void* const* d_in, const int* in_sizes, int n_in,
                              void* d_out, int out_size, void* d_ws, size_t ws_size,
                              hipStream_t stream) {
    const float* emb    = (const float*)d_in[0];
    const float* Wk     = (const float*)d_in[1];
    const float* bk     = (const float*)d_in[2];
    const float* Wq     = (const float*)d_in[3];
    const float* bq     = (const float*)d_in[4];
    const float* belief = (const float*)d_in[5];
    const int*   idcs   = (const int*)d_in[6];
    float* out          = (float*)d_out;

    const int S = in_sizes[0] / EMB;                 // 50000
    const int numTiles = (S + TILE - 1) / TILE;      // 391

    char* ws = (char*)d_ws;
    size_t off = 0;
    auto align256 = [](size_t x) { return (x + 255) & ~(size_t)255; };

    _Float16* keysF16 = (_Float16*)(ws + off); off += align256((size_t)S * HID * sizeof(_Float16));
    _Float16* qF16    = (_Float16*)(ws + off); off += align256((size_t)BATCH * KSP * HID * sizeof(_Float16));
    float* pmax   = (float*)(ws + off); off += align256((size_t)BATCH * KSP * numTiles * sizeof(float));
    float* psum   = (float*)(ws + off); off += align256((size_t)BATCH * KSP * numTiles * sizeof(float));
    float* rowmax = (float*)(ws + off); off += align256((size_t)BATCH * KSP * sizeof(float));
    float* roww   = (float*)(ws + off); off += align256((size_t)BATCH * KSP * sizeof(float));
    (void)ws_size; (void)n_in; (void)out_size;

    k_keys<<<numTiles, 256, 0, stream>>>(emb, Wk, bk, keysF16, S);
    k_queries<<<(BATCH * KSP) / TILE, 256, 0, stream>>>(emb, Wq, bq, idcs, qF16);
    k_stats<<<dim3(numTiles, BATCH), 256, 0, stream>>>(qF16, keysF16, pmax, psum, S, numTiles);
    k_reduce<<<(BATCH * KSP) / 256, 256, 0, stream>>>(pmax, psum, belief, rowmax, roww, numTiles);
    k_out<<<dim3(numTiles, BATCH), 256, 0, stream>>>(qF16, keysF16, rowmax, roww, out, S, numTiles);
}
